// MessageLayer_45561013076244
// MI455X (gfx1250) — matrix operand order repaired and checked by reading, not compiled, not run on hardware
//
#include <hip/hip_runtime.h>
#include <hip/hip_bf16.h>
#include <math.h>
#include <stdint.h>

#define N_ATOMS  20000
#define N_EDGES  640000
#define FDIM     128
#define F3       384
#define N_RADIAL 20
#define CUTOFF   5.0f
#define PI_F     3.14159265358979f
#define EPB      8            // edges per block in the edge kernel

typedef __attribute__((ext_vector_type(16))) _Float16 v16h;
typedef __attribute__((ext_vector_type(8)))  _Float16 v8h;
typedef __attribute__((ext_vector_type(8)))  float    v8f;

// ---------------------------------------------------------------------------
// Kernel 0: seed outputs (out = [q | mu]) and convert q -> f16 workspace.
// ---------------------------------------------------------------------------
__global__ __launch_bounds__(256)
void painn_init_out(const float* __restrict__ q,
                    const float* __restrict__ mu,
                    float* __restrict__ out,
                    _Float16* __restrict__ q16) {
  size_t i  = (size_t)blockIdx.x * blockDim.x + threadIdx.x;
  size_t nq = (size_t)N_ATOMS * FDIM;
  size_t nt = nq + (size_t)N_ATOMS * 3 * FDIM;
  if (i < nq) {
    float v = q[i];
    out[i] = v;
    q16[i] = (_Float16)v;
  } else if (i < nt) {
    out[i] = mu[i - nq];
  }
}

// ---------------------------------------------------------------------------
// Kernel P: pack a [128 x N] f32 weight matrix into f16 WMMA-B fragments.
// Fragment order: [ct][kt][lane][16 halves]; lane's halves are K-contiguous
// as A loads them (lanes 16-31 hold K+16). One thread per (ct,kt,lane).
// ---------------------------------------------------------------------------
__global__ __launch_bounds__(256)
void painn_pack_w(const float* __restrict__ W,
                  _Float16* __restrict__ dst,
                  int ncols, int total_frag_lanes) {
  const int t = blockIdx.x * blockDim.x + threadIdx.x;
  if (t >= total_frag_lanes) return;
  const int lane = t & 31;
  const int kt   = (t >> 5) & 3;           // 128/32 = 4 K-tiles
  const int ct   = t >> 7;                 // column tile
  const int col  = ct * 16 + (lane & 15);
  const int koff = kt * 32 + ((lane < 16) ? 0 : 16);
  _Float16* d = dst + (size_t)t * 16;
  #pragma unroll
  for (int h = 0; h < 16; ++h)
    d[h] = (_Float16)W[(size_t)(koff + h) * ncols + col];
}

// ---------------------------------------------------------------------------
// Shared WMMA tile body: A tile (16x128 f16) async-staged to LDS; B from
// pre-packed fragments; one wave -> one 16x16 f32 tile.
// ---------------------------------------------------------------------------
__device__ __forceinline__ v8f wmma_tile_from_lds(const _Float16* sA_row,
                                                  const _Float16* bfrag_base,
                                                  int lane) {
  const int a_koff = (lane < 16) ? 0 : 16;
  v8f c = {};
  #pragma unroll
  for (int kt = 0; kt < 4; ++kt) {
    const int k0 = kt * 32;
    v8h lo = *(const v8h*)(sA_row + k0 + a_koff);
    v8h hi = *(const v8h*)(sA_row + k0 + a_koff + 8);
    v16h a;
    #pragma unroll
    for (int h = 0; h < 8; ++h) { a[h] = lo[h]; a[h + 8] = hi[h]; }
    v16h b = *(const v16h*)(bfrag_base + ((size_t)kt * 32 + lane) * 16);
    c = __builtin_amdgcn_wmma_f32_16x16x32_f16(false, a, false, b,
                                               (short)0, c, false, false);
  }
  return c;
}

__device__ __forceinline__ void async_stage_tile(const _Float16* gsrc,
                                                 _Float16* sA, int tid) {
  // 16 rows x 128 halves = 4096B; 256 threads x 16B each, ASYNCcnt-tracked.
  unsigned lds_off = (unsigned)(uintptr_t)(&sA[0]) + (unsigned)tid * 16u;
  unsigned long long ga =
      (unsigned long long)(uintptr_t)gsrc + (unsigned long long)tid * 16u;
  asm volatile("global_load_async_to_lds_b128 %0, %1, off"
               :: "v"(lds_off), "v"(ga) : "memory");
  asm volatile("s_wait_asynccnt 0x0" ::: "memory");
  __syncthreads();
}

// ---------------------------------------------------------------------------
// Kernel 1a: h = silu(q @ W1 + b1)  -> f16 workspace
// ---------------------------------------------------------------------------
__global__ __launch_bounds__(256)
void painn_mlp1(const _Float16* __restrict__ q16,
                const _Float16* __restrict__ w1p,
                const float* __restrict__ b1,
                _Float16* __restrict__ hout) {
  __shared__ __align__(16) _Float16 sA[16 * FDIM];
  const int tid = threadIdx.x, wave = tid >> 5, lane = tid & 31;
  const int row_base = blockIdx.x * 16;

  async_stage_tile(q16 + (size_t)row_base * FDIM, sA, tid);

  const _Float16* arow = &sA[(lane & 15) * FDIM];
  v8f c = wmma_tile_from_lds(arow, w1p + (size_t)wave * 4 * 32 * 16, lane);

  const int col   = wave * 16 + (lane & 15);
  const float bias = b1[col];
  const int m_off = (lane < 16) ? 0 : 8;
  #pragma unroll
  for (int r = 0; r < 8; ++r) {
    const int m = row_base + r + m_off;
    const float v = c[r] + bias;
    // SiLU with fast reciprocal (v_rcp_f32): v * rcp(1 + exp(-v))
    const float s = v * __builtin_amdgcn_rcpf(1.0f + __expf(-v));
    hout[(size_t)m * FDIM + col] = (_Float16)s;
  }
}

// ---------------------------------------------------------------------------
// Kernel 1b: x = h @ W2 + b2  -> f32 workspace  (grid.y = 3 -> 24 col tiles)
// ---------------------------------------------------------------------------
__global__ __launch_bounds__(256)
void painn_mlp2(const _Float16* __restrict__ hin,
                const _Float16* __restrict__ w2p,
                const float* __restrict__ b2,
                float* __restrict__ xout) {
  __shared__ __align__(16) _Float16 sA[16 * FDIM];
  const int tid = threadIdx.x, wave = tid >> 5, lane = tid & 31;
  const int row_base = blockIdx.x * 16;
  const int ct = blockIdx.y * 8 + wave;

  async_stage_tile(hin + (size_t)row_base * FDIM, sA, tid);

  const _Float16* arow = &sA[(lane & 15) * FDIM];
  v8f c = wmma_tile_from_lds(arow, w2p + (size_t)ct * 4 * 32 * 16, lane);

  const int col   = ct * 16 + (lane & 15);
  const float bias = b2[col];
  const int m_off = (lane < 16) ? 0 : 8;
  #pragma unroll
  for (int r = 0; r < 8; ++r) {
    const int m = row_base + r + m_off;
    xout[(size_t)m * F3 + col] = c[r] + bias;
  }
}

// ---------------------------------------------------------------------------
// Kernel 2: edges. 8 edges/block; Wf+bf and per-edge basis staged in LDS.
// 4 rounds of (2 edges x 128 channels), 4 f32 atomics per (edge,channel).
// Scatter target (41 MB) is resident in MI455X's 192 MB L2.
// ---------------------------------------------------------------------------
__global__ __launch_bounds__(256)
void painn_edge(const int* __restrict__ edge_index,
                const float* __restrict__ ew,
                const float* __restrict__ Wf,
                const float* __restrict__ bf,
                const float* __restrict__ x,
                const float* __restrict__ mu,
                float* __restrict__ q_out,
                float* __restrict__ mu_out) {
  __shared__ __align__(16) float sWf[N_RADIAL * F3];  // 30720 B
  __shared__ __align__(16) float sbf[F3];
  __shared__ float sbasis[EPB][N_RADIAL];
  __shared__ float sdir[EPB][3];
  __shared__ float senv[EPB];
  __shared__ int   sij[EPB][2];

  const int tid = threadIdx.x;
  const int e0  = blockIdx.x * EPB;

  // Stage filter weights (vectorized b128 -> ds_store)
  for (int idx = tid; idx < (N_RADIAL * F3) / 4; idx += 256)
    ((float4*)sWf)[idx] = ((const float4*)Wf)[idx];
  for (int idx = tid; idx < F3 / 4; idx += 256)
    ((float4*)sbf)[idx] = ((const float4*)bf)[idx];

  // Per-edge scalars (8 threads) and basis (8*20 threads)
  if (tid < EPB) {
    const int e = e0 + tid;
    const float wx = ew[(size_t)e * 3 + 0];
    const float wy = ew[(size_t)e * 3 + 1];
    const float wz = ew[(size_t)e * 3 + 2];
    const float d  = sqrtf(wx * wx + wy * wy + wz * wz);
    const float inv_d = __builtin_amdgcn_rcpf(d);
    sdir[tid][0] = wx * inv_d;
    sdir[tid][1] = wy * inv_d;
    sdir[tid][2] = wz * inv_d;
    senv[tid] = (d < CUTOFF) ? 0.5f * (__cosf(d * (PI_F / CUTOFF)) + 1.0f) : 0.0f;
    sij[tid][0] = edge_index[e];
    sij[tid][1] = edge_index[N_EDGES + e];
  }
  if (tid < EPB * N_RADIAL) {
    const int el = tid / N_RADIAL, r = tid % N_RADIAL;
    const int e = e0 + el;
    const float wx = ew[(size_t)e * 3 + 0];
    const float wy = ew[(size_t)e * 3 + 1];
    const float wz = ew[(size_t)e * 3 + 2];
    const float d  = sqrtf(wx * wx + wy * wy + wz * wz);
    sbasis[el][r] = __sinf(d * ((float)(r + 1) * (PI_F / CUTOFF))) *
                    __builtin_amdgcn_rcpf(d);
  }
  __syncthreads();

  const int f = tid & 127;
  #pragma unroll
  for (int round = 0; round < EPB / 2; ++round) {
    const int el = round * 2 + (tid >> 7);
    const int i = sij[el][0];
    const int j = sij[el][1];
    const float env = senv[el];

    float filt[3];
    #pragma unroll
    for (int t = 0; t < 3; ++t) {
      const int cch = t * FDIM + f;
      float acc = sbf[cch];
      #pragma unroll
      for (int r = 0; r < N_RADIAL; ++r)
        acc += sbasis[el][r] * sWf[r * F3 + cch];
      filt[t] = acc * env;
    }

    const size_t xb   = (size_t)j * F3;
    const float dq    = filt[0] * x[xb + f];
    const float dmuR  = filt[1] * x[xb + FDIM + f];
    const float dmumu = filt[2] * x[xb + 2 * FDIM + f];

    atomicAdd(&q_out[(size_t)i * FDIM + f], dq);

    const size_t mb_j = (size_t)j * 3 * FDIM + f;
    const size_t mb_i = (size_t)i * 3 * FDIM + f;
    atomicAdd(&mu_out[mb_i],            dmuR * sdir[el][0] + dmumu * mu[mb_j]);
    atomicAdd(&mu_out[mb_i + FDIM],     dmuR * sdir[el][1] + dmumu * mu[mb_j + FDIM]);
    atomicAdd(&mu_out[mb_i + 2 * FDIM], dmuR * sdir[el][2] + dmumu * mu[mb_j + 2 * FDIM]);
  }
}

// ---------------------------------------------------------------------------
extern "C" void kernel_launch(void* const* d_in, const int* in_sizes, int n_in,
                              void* d_out, int out_size, void* d_ws, size_t ws_size,
                              hipStream_t stream) {
  const float* q   = (const float*)d_in[0];
  const float* mu  = (const float*)d_in[1];
  const int*   ei  = (const int*)  d_in[2];
  const float* ew  = (const float*)d_in[3];
  const float* W1  = (const float*)d_in[4];
  const float* b1  = (const float*)d_in[5];
  const float* W2  = (const float*)d_in[6];
  const float* b2  = (const float*)d_in[7];
  const float* Wf  = (const float*)d_in[8];
  const float* bf  = (const float*)d_in[9];
  (void)in_sizes; (void)n_in; (void)out_size; (void)ws_size;

  float* q_out  = (float*)d_out;
  float* mu_out = q_out + (size_t)N_ATOMS * FDIM;

  // Workspace layout (bytes): q16 | h16 | x_f32 | w1pack | w2pack
  char* ws = (char*)d_ws;
  _Float16* q16  = (_Float16*)(ws);                              // 5,120,000
  _Float16* h16  = (_Float16*)(ws + 5120000);                    // 5,120,000
  float*    x_ws = (float*)   (ws + 10240000);                   // 30,720,000
  _Float16* w1p  = (_Float16*)(ws + 40960000);                   // 32,768
  _Float16* w2p  = (_Float16*)(ws + 40992768);                   // 98,304

  // 0) out = [q | mu], q -> f16
  painn_init_out<<<(N_ATOMS * 512 + 255) / 256, 256, 0, stream>>>(q, mu,
                                                                  (float*)d_out, q16);
  // P) pack weights into WMMA-B fragment order
  painn_pack_w<<<(8 * 4 * 32 + 255) / 256, 256, 0, stream>>>(W1, w1p, FDIM,
                                                             8 * 4 * 32);
  painn_pack_w<<<(24 * 4 * 32 + 255) / 256, 256, 0, stream>>>(W2, w2p, F3,
                                                              24 * 4 * 32);
  // 1a) h = silu(q@W1+b1)
  painn_mlp1<<<N_ATOMS / 16, 256, 0, stream>>>(q16, w1p, b1, h16);
  // 1b) x = h@W2+b2
  painn_mlp2<<<dim3(N_ATOMS / 16, 3), 256, 0, stream>>>(h16, w2p, b2, x_ws);
  // 2) edge filter + gather + scatter-add
  painn_edge<<<N_EDGES / EPB, 256, 0, stream>>>(ei, ew, Wf, bf, x_ws, mu,
                                                q_out, mu_out);
}
